// PointNetCls_77506979823777
// MI455X (gfx1250) — matrix operand order repaired and checked by reading, not compiled, not run on hardware
//
#include <hip/hip_runtime.h>
#include <hip/hip_bf16.h>
#include <math.h>

// ---------------- types / constants ----------------
typedef __attribute__((ext_vector_type(16))) __bf16 v16bf;
typedef __attribute__((ext_vector_type(8)))  __bf16 v8bf;
typedef __attribute__((ext_vector_type(8)))  float  v8f;
typedef __attribute__((ext_vector_type(4)))  float  v4f;

#define BB 16
#define NP 16384
#define VV 1000
#define TP 64   // points per block tile

// ---------------- small device helpers ----------------
__device__ __forceinline__ float fast_tanh(float x) {
#if __has_builtin(__builtin_amdgcn_tanhf)
  return __builtin_amdgcn_tanhf(x);   // CDNA5 V_TANH_F32 (TRANS op, co-executes with WMMA)
#else
  return tanhf(x);
#endif
}
__device__ __forceinline__ float gelu_f(float x) {
  // tanh-form GELU; branchless
  float u = 0.7978845608028654f * x * (1.0f + 0.044715f * x * x);
  return 0.5f * x * (1.0f + fast_tanh(u));
}
// monotonic float<->uint encoding so atomicMax/Min(u32) == float max/min
__device__ __forceinline__ unsigned fenc(float f) {
  unsigned u = __float_as_uint(f);
  return (u & 0x80000000u) ? ~u : (u | 0x80000000u);
}
__device__ __forceinline__ float fdec(unsigned e) {
  unsigned u = (e & 0x80000000u) ? (e & 0x7FFFFFFFu) : ~e;
  return __uint_as_float(u);
}

// ---------------- fused conv1x1 stage over a TP-point tile ----------------
// Wg  : folded bf16 weights [COUT][CINP] row-major in global (256B aligned)
// Xin : LDS activations point-major [TP][CINP]  -> B fragment = 32 contiguous bytes/lane
// OUTBF: write LDS point-major [TP][COUT] bf16 (16B contiguous store per lane)
// OUTF : write LDS channel-major [COUT][TP] f32 (feeds reductions)
// All behavior flags are compile-time -> no branches in the hot loop.
template<int CINP, int COUT, int TPP, bool HASB, bool ACT, bool OUTBF, bool OUTF>
__device__ __forceinline__ void conv_stage(const __bf16* __restrict__ Wg,
                                           const float* __restrict__ bias,
                                           const __bf16* Xin, __bf16* XoutBf, float* XoutF,
                                           int lane, int wave) {
  constexpr int MT = COUT / 16, NT = TPP / 16, KT = CINP / 32, NW = 8;
  for (int t = wave; t < MT * NT; t += NW) {  // uniform per-wave -> EXEC stays all-1s
    int m0 = (t % MT) * 16, n0 = (t / MT) * 16;
    // C/D layout: VGPR r -> M = m0 + r + (laneHi?8:0), N = n0 + (lane&15)
    int nn = n0 + (lane & 15);
    int mb = m0 + ((lane >> 4) << 3);
    v4f b0 = {}, b1 = {};
    if constexpr (HASB) {            // 8 contiguous floats, 16B-aligned -> b128 x2, once/tile
      b0 = *(const v4f*)(bias + mb);
      b1 = *(const v4f*)(bias + mb + 4);
    }
    v8f acc = {};
#pragma unroll
    for (int kb = 0; kb < KT; ++kb) {
      // A fragment: lane row M=m0+(lane&15); K = kb*32 + (laneHi?16:0) + {0..15}
      const __bf16* row = Wg + (size_t)(m0 + (lane & 15)) * CINP + kb * 32 + ((lane >> 4) << 4);
      v8bf lo = *(const v8bf*)row;          // 16B
      v8bf hi = *(const v8bf*)(row + 8);    // 16B
      v16bf a;
#pragma unroll
      for (int e = 0; e < 8; ++e) { a[e] = lo[e]; a[8 + e] = hi[e]; }
      // B fragment: lane column N=n0+(lane&15); K = kb*32 + (laneHi?16:0) + {0..15}
      const __bf16* bp = Xin + (size_t)(n0 + (lane & 15)) * CINP + kb * 32 + ((lane >> 4) << 4);
      v16bf bm = *(const v16bf*)bp;         // 32B contiguous -> ds_load_b128 x2
      acc = __builtin_amdgcn_wmma_f32_16x16x32_bf16(false, a, false, bm, (short)0, acc,
                                                    false, false);
    }
    float vv[8];
#pragma unroll
    for (int r = 0; r < 8; ++r) {
      float v = acc[r];
      if constexpr (HASB) v += (r < 4) ? b0[r] : b1[r - 4];
      if constexpr (ACT) v = gelu_f(v);
      vv[r] = v;
    }
    if constexpr (OUTBF) {
      v8bf o;
#pragma unroll
      for (int r = 0; r < 8; ++r) o[r] = (__bf16)vv[r];
      *(v8bf*)(XoutBf + (size_t)nn * COUT + mb) = o;  // 16B contiguous store
    }
    if constexpr (OUTF) {
#pragma unroll
      for (int r = 0; r < 8; ++r) XoutF[(mb + r) * TPP + nn] = vv[r];
    }
  }
}

// ---------------- K0: fold BN into bf16 weights ----------------
__global__ void k_fold(const float* __restrict__ W, const float* __restrict__ g,
                       const float* __restrict__ bb, const float* __restrict__ m,
                       const float* __restrict__ v, __bf16* __restrict__ Wout,
                       float* __restrict__ biasOut, int O, int CIN, int CINP) {
  int i = blockIdx.x * blockDim.x + threadIdx.x;
  if (i >= O * CINP) return;
  int o = i / CINP, k = i % CINP;
  float s = g ? g[o] * rsqrtf(v[o] + 1e-5f) : 1.0f;
  float w = (k < CIN) ? W[o * CIN + k] * s : 0.0f;
  Wout[i] = (__bf16)w;
  if (k == 0 && biasOut) biasOut[o] = g ? (bb[o] - s * m[o]) : 0.0f;
}

// ---------------- K1: STN3 conv chain (3->64->128->256) + channel max ----------------
__global__ void __launch_bounds__(256) k_stn3(const float* __restrict__ xyz,
                                              const __bf16* __restrict__ W1,
                                              const float* __restrict__ B1,
                                              const __bf16* __restrict__ W2,
                                              const float* __restrict__ B2,
                                              const __bf16* __restrict__ W3,
                                              unsigned* __restrict__ g3max) {
  int b = blockIdx.y, n0g = blockIdx.x * TP;
  int tid = threadIdx.x, lane = tid & 31, wave = tid >> 5;
  __shared__ __align__(32) __bf16 X0[TP * 32];
  __shared__ __align__(32) __bf16 X1[TP * 64];
  __shared__ __align__(32) __bf16 X2[TP * 128];
  __shared__ __align__(16) float  X3[256 * TP];
  if (tid < TP) {
    int n = tid;
    size_t pb = ((size_t)b * NP + (n0g + n)) * 3;
    v16bf r;
#pragma unroll
    for (int e = 0; e < 16; ++e) r[e] = (__bf16)0.0f;
    r[0] = (__bf16)xyz[pb];
    r[1] = (__bf16)xyz[pb + 1];
    r[2] = (__bf16)xyz[pb + 2];
    *(v16bf*)(X0 + n * 32) = r;
    v16bf z;
#pragma unroll
    for (int e = 0; e < 16; ++e) z[e] = (__bf16)0.0f;
    *(v16bf*)(X0 + n * 32 + 16) = z;
  }
  __syncthreads();
  conv_stage<32, 64, TP, true, true, true, false>(W1, B1, X0, X1, nullptr, lane, wave);
  __syncthreads();
  conv_stage<64, 128, TP, true, true, true, false>(W2, B2, X1, X2, nullptr, lane, wave);
  __syncthreads();
  conv_stage<128, 256, TP, false, false, false, true>(W3, nullptr, X2, nullptr, X3, lane, wave);
  __syncthreads();
  for (int c = tid; c < 256; c += 256) {
    float mx = -3.4e38f;
    for (int n = 0; n < TP; ++n) mx = fmaxf(mx, X3[c * TP + n]);
    atomicMax(&g3max[b * 256 + c], fenc(mx));
  }
}

// ---------------- K2: STN FC head: gelu(max)->l1->l2->l3+bias ----------------
__global__ void __launch_bounds__(256) k_fc(const unsigned* __restrict__ gmaxE,
                                            const float* __restrict__ l1,
                                            const float* __restrict__ l2,
                                            const float* __restrict__ l3,
                                            const float* __restrict__ l3b,
                                            float* __restrict__ ToutF,
                                            __bf16* __restrict__ ToutBf, int OD) {
  int b = blockIdx.x, tid = threadIdx.x;
  __shared__ float h0[256], h1[128], h2[64];
  h0[tid] = gelu_f(fdec(gmaxE[b * 256 + tid]));
  __syncthreads();
  if (tid < 128) { float s = 0; for (int k = 0; k < 256; ++k) s += l1[tid * 256 + k] * h0[k]; h1[tid] = gelu_f(s); }
  __syncthreads();
  if (tid < 64)  { float s = 0; for (int k = 0; k < 128; ++k) s += l2[tid * 128 + k] * h1[k]; h2[tid] = gelu_f(s); }
  __syncthreads();
  for (int o = tid; o < OD; o += 256) {
    float s = l3b[o];
    for (int k = 0; k < 64; ++k) s += l3[o * 64 + k] * h2[k];
    if (ToutF)  ToutF[(size_t)b * OD + o]  = s;
    if (ToutBf) ToutBf[(size_t)b * OD + o] = (__bf16)s;
  }
}

// ---------------- K3: T3 apply + c1a/c1b + STNnd conv chain + bounds ----------------
__global__ void __launch_bounds__(256) k_feat(const float* __restrict__ xyz,
    const float* __restrict__ features, const float* __restrict__ T3,
    const __bf16* __restrict__ Wc1a, const float* __restrict__ Bc1a,
    const __bf16* __restrict__ Wc1b, const float* __restrict__ Bc1b,
    const __bf16* __restrict__ Wn1, const float* __restrict__ Bn1,
    const __bf16* __restrict__ Wn2, const float* __restrict__ Bn2,
    const __bf16* __restrict__ Wn3, float* __restrict__ xyzp, __bf16* __restrict__ x64pm,
    unsigned* __restrict__ bminE, unsigned* __restrict__ bmaxE,
    unsigned* __restrict__ gndmax) {
  int b = blockIdx.y, n0g = blockIdx.x * TP;
  int tid = threadIdx.x, lane = tid & 31, wave = tid >> 5;
  __shared__ __align__(32) __bf16 X0[TP * 32];
  __shared__ __align__(32) __bf16 X1[TP * 64];
  __shared__ __align__(32) __bf16 X2[TP * 64];
  __shared__ __align__(32) __bf16 X3[TP * 64];
  __shared__ __align__(32) __bf16 X4[TP * 128];
  __shared__ __align__(16) float  X5[256 * TP];
  __shared__ float t3[9];
  __shared__ unsigned smn[3], smx[3];
  if (tid < 9) t3[tid] = T3[b * 9 + tid];
  if (tid < 3) { smn[tid] = 0xFFFFFFFFu; smx[tid] = 0u; }
  __syncthreads();
  if (tid < TP) {
    int n = tid;
    size_t pb = ((size_t)b * NP + (n0g + n)) * 3;
    float x = xyz[pb], y = xyz[pb + 1], z = xyz[pb + 2];
    float p[3];
#pragma unroll
    for (int d = 0; d < 3; ++d) {
      p[d] = t3[d * 3 + 0] * x + t3[d * 3 + 1] * y + t3[d * 3 + 2] * z;
      xyzp[((size_t)b * 3 + d) * NP + n0g + n] = p[d];
      unsigned e = fenc(p[d]);
      atomicMin(&smn[d], e);
      atomicMax(&smx[d], e);
    }
    v16bf r;
#pragma unroll
    for (int e = 0; e < 16; ++e) r[e] = (__bf16)0.0f;
    r[0] = (__bf16)p[0]; r[1] = (__bf16)p[1]; r[2] = (__bf16)p[2];
#pragma unroll
    for (int c = 3; c < 6; ++c)
      r[c] = (__bf16)features[((size_t)b * 6 + c) * NP + n0g + n];
    *(v16bf*)(X0 + n * 32) = r;
    v16bf zz;
#pragma unroll
    for (int e = 0; e < 16; ++e) zz[e] = (__bf16)0.0f;
    *(v16bf*)(X0 + n * 32 + 16) = zz;
  }
  __syncthreads();
  if (tid < 3) { atomicMin(&bminE[b * 3 + tid], smn[tid]); atomicMax(&bmaxE[b * 3 + tid], smx[tid]); }
  conv_stage<32, 64, TP, true, true, true, false>(Wc1a, Bc1a, X0, X1, nullptr, lane, wave);
  __syncthreads();
  conv_stage<64, 64, TP, true, true, true, false>(Wc1b, Bc1b, X1, X2, nullptr, lane, wave);
  __syncthreads();
  {  // stash 64-ch features point-major to global (contiguous 16B copies)
    const uint4* src = (const uint4*)X2;
    uint4* dst = (uint4*)(x64pm + ((size_t)b * NP + n0g) * 64);
    for (int i = tid; i < TP * 64 * 2 / 16; i += 256) dst[i] = src[i];
  }
  conv_stage<64, 64, TP, true, true, true, false>(Wn1, Bn1, X2, X3, nullptr, lane, wave);
  __syncthreads();
  conv_stage<64, 128, TP, true, true, true, false>(Wn2, Bn2, X3, X4, nullptr, lane, wave);
  __syncthreads();
  conv_stage<128, 256, TP, false, false, false, true>(Wn3, nullptr, X4, nullptr, X5, lane, wave);
  __syncthreads();
  for (int c = tid; c < 256; c += 256) {
    float mx = -3.4e38f;
    for (int n = 0; n < TP; ++n) mx = fmaxf(mx, X5[c * TP + n]);
    atomicMax(&gndmax[b * 256 + c], fenc(mx));
  }
}

// ---------------- K5: voxel index + counts ----------------
__global__ void __launch_bounds__(256) k_voxidx(const float* __restrict__ xyzp,
                                                const unsigned* __restrict__ bminE,
                                                const unsigned* __restrict__ bmaxE,
                                                int* __restrict__ idx_out,
                                                int* __restrict__ counts) {
  int b = blockIdx.y, n = blockIdx.x * 256 + threadIdx.x;
  __shared__ float mn[3], rng[3];
  if (threadIdx.x < 3) {
    float lo = fdec(bminE[b * 3 + threadIdx.x]);
    float hi = fdec(bmaxE[b * 3 + threadIdx.x]);
    mn[threadIdx.x] = lo; rng[threadIdx.x] = hi - lo + 1e-8f;
  }
  __syncthreads();
  int id = 0;
  for (int d = 0; d < 3; ++d) {
    float v = xyzp[((size_t)b * 3 + d) * NP + n];
    float nrm = fminf(fmaxf((v - mn[d]) / rng[d], 0.0f), 1.0f - 1e-6f);
    int q = (int)floorf(nrm * 10.0f);
    q = q < 0 ? 0 : (q > 9 ? 9 : q);
    id = id * 10 + q;
  }
  idx_out[(size_t)b * NP + n] = id;
  atomicAdd(&counts[b * VV + id], 1);
}

// ---------------- K6: Tn apply + c2a/c2b/c2c + voxel atomic max ----------------
__global__ void __launch_bounds__(256) k_chain2(const __bf16* __restrict__ x64pm,
    const __bf16* __restrict__ TnBf, const __bf16* __restrict__ Wc2a,
    const float* __restrict__ Bc2a, const __bf16* __restrict__ Wc2b,
    const float* __restrict__ Bc2b, const __bf16* __restrict__ Wc2c,
    const float* __restrict__ Bc2c, const int* __restrict__ idx,
    unsigned* __restrict__ voxE) {
  int b = blockIdx.y, n0g = blockIdx.x * TP;
  int tid = threadIdx.x, lane = tid & 31, wave = tid >> 5;
  __shared__ __align__(32) __bf16 X0[TP * 64];
  __shared__ __align__(32) __bf16 X1[TP * 64];
  __shared__ __align__(32) __bf16 X2[TP * 64];
  __shared__ __align__(32) __bf16 X3[TP * 128];
  __shared__ __align__(16) float  X4[256 * TP];
  __shared__ int vid[TP];
  {  // point-major global -> point-major LDS: pure 16B copies
    const uint4* src = (const uint4*)(x64pm + ((size_t)b * NP + n0g) * 64);
    uint4* dst = (uint4*)X0;
    for (int i = tid; i < TP * 64 * 2 / 16; i += 256) dst[i] = src[i];
  }
  if (tid < TP) vid[tid] = idx[(size_t)b * NP + n0g + tid];
  __syncthreads();
  conv_stage<64, 64, TP, false, false, true, false>(TnBf + (size_t)b * 4096, nullptr, X0, X1,
                                                    nullptr, lane, wave);
  __syncthreads();
  conv_stage<64, 64, TP, true, true, true, false>(Wc2a, Bc2a, X1, X2, nullptr, lane, wave);
  __syncthreads();
  conv_stage<64, 128, TP, true, true, true, false>(Wc2b, Bc2b, X2, X3, nullptr, lane, wave);
  __syncthreads();
  conv_stage<128, 256, TP, true, false, false, true>(Wc2c, Bc2c, X3, nullptr, X4, lane, wave);
  __syncthreads();
  for (int i = tid; i < 256 * TP; i += 256) {
    int c = i / TP, n = i % TP;
    atomicMax(&voxE[((size_t)b * 256 + c) * VV + vid[n]], fenc(X4[i]));
  }
}

// ---------------- K7: global_feat + logits ----------------
__global__ void __launch_bounds__(256) k_final1(const unsigned* __restrict__ voxE,
                                                const int* __restrict__ counts,
                                                const float* __restrict__ head,
                                                float* __restrict__ logits,
                                                float* __restrict__ gfeat) {
  int b = blockIdx.x, c = threadIdx.x;
  __shared__ float gf[256];
  float s = 0.0f;
  for (int v = 0; v < VV; ++v)
    if (counts[b * VV + v] > 0) s += fdec(voxE[((size_t)b * 256 + c) * VV + v]);
  s *= (1.0f / (float)VV);
  gf[c] = s;
  gfeat[b * 256 + c] = s;
  __syncthreads();
  if (c < 40) {
    float a = 0.0f;
    for (int k = 0; k < 256; ++k) a += head[c * 256 + k] * gf[k];
    logits[b * 40 + c] = a;
  }
}

// ---------------- K8: voxel activation (L2 norm over channels) ----------------
__global__ void __launch_bounds__(256) k_vact(const unsigned* __restrict__ voxE,
                                              const int* __restrict__ counts,
                                              float* __restrict__ vact) {
  int g = blockIdx.x * blockDim.x + threadIdx.x;
  if (g >= BB * VV) return;
  int b = g / VV, v = g % VV;
  float s = 0.0f;
  if (counts[b * VV + v] > 0)
    for (int c = 0; c < 256; ++c) {
      float f = fdec(voxE[((size_t)b * 256 + c) * VV + v]);
      s += f * f;
    }
  vact[g] = sqrtf(s);
}

// ---------------- host side ----------------
struct Ptrs {
  const float *features, *xyz;
  const float *s3w1, *s3w2, *s3w3, *s3l1, *s3l2, *s3l3, *s3l3b;
  const float *s3g1, *s3bb1, *s3m1, *s3v1, *s3g2, *s3bb2, *s3m2, *s3v2;
  const float *ndw1, *ndw2, *ndw3, *ndl1, *ndl2, *ndl3, *ndl3b;
  const float *ndg1, *ndbb1, *ndm1, *ndv1, *ndg2, *ndbb2, *ndm2, *ndv2;
  const float *c1aw, *c1ag, *c1abb, *c1am, *c1av;
  const float *c1bw, *c1bg, *c1bbb, *c1bm, *c1bv;
  const float *c2aw, *c2ag, *c2abb, *c2am, *c2av;
  const float *c2bw, *c2bg, *c2bbb, *c2bm, *c2bv;
  const float *c2cw, *c2cg, *c2cbb, *c2cm, *c2cv;
  const float *head;
};

static void resolve(void* const* d_in, const int* in_sizes, int n_in, Ptrs& P) {
  auto F = [&](int i) { return (const float*)d_in[i]; };
  bool insertion = (n_in > 1 && in_sizes[1] == BB * NP * 3);  // xyz second -> dict insertion order
  if (insertion) {
    P.features = F(0); P.xyz = F(1);
    P.s3w1 = F(2); P.s3g1 = F(3); P.s3bb1 = F(4); P.s3m1 = F(5); P.s3v1 = F(6);
    P.s3w2 = F(7); P.s3g2 = F(8); P.s3bb2 = F(9); P.s3m2 = F(10); P.s3v2 = F(11);
    P.s3w3 = F(12); P.s3l1 = F(13); P.s3l2 = F(14); P.s3l3 = F(15); P.s3l3b = F(16);
    P.ndw1 = F(17); P.ndg1 = F(18); P.ndbb1 = F(19); P.ndm1 = F(20); P.ndv1 = F(21);
    P.ndw2 = F(22); P.ndg2 = F(23); P.ndbb2 = F(24); P.ndm2 = F(25); P.ndv2 = F(26);
    P.ndw3 = F(27); P.ndl1 = F(28); P.ndl2 = F(29); P.ndl3 = F(30); P.ndl3b = F(31);
    P.c1aw = F(32); P.c1ag = F(33); P.c1abb = F(34); P.c1am = F(35); P.c1av = F(36);
    P.c1bw = F(37); P.c1bg = F(38); P.c1bbb = F(39); P.c1bm = F(40); P.c1bv = F(41);
    P.c2aw = F(42); P.c2ag = F(43); P.c2abb = F(44); P.c2am = F(45); P.c2av = F(46);
    P.c2bw = F(47); P.c2bg = F(48); P.c2bbb = F(49); P.c2bm = F(50); P.c2bv = F(51);
    P.c2cw = F(52); P.c2cg = F(53); P.c2cbb = F(54); P.c2cm = F(55); P.c2cv = F(56);
    P.head = F(57);
  } else {  // jax tree (sorted keys): bn leaves b,g,m,v; xyz last
    P.features = F(0);
    P.c1aw = F(1);  P.c1abb = F(2);  P.c1ag = F(3);  P.c1am = F(4);  P.c1av = F(5);
    P.c1bw = F(6);  P.c1bbb = F(7);  P.c1bg = F(8);  P.c1bm = F(9);  P.c1bv = F(10);
    P.c2aw = F(11); P.c2abb = F(12); P.c2ag = F(13); P.c2am = F(14); P.c2av = F(15);
    P.c2bw = F(16); P.c2bbb = F(17); P.c2bg = F(18); P.c2bm = F(19); P.c2bv = F(20);
    P.c2cw = F(21); P.c2cbb = F(22); P.c2cg = F(23); P.c2cm = F(24); P.c2cv = F(25);
    P.head = F(26);
    P.s3bb1 = F(27); P.s3g1 = F(28); P.s3m1 = F(29); P.s3v1 = F(30);
    P.s3bb2 = F(31); P.s3g2 = F(32); P.s3m2 = F(33); P.s3v2 = F(34);
    P.s3l1 = F(35); P.s3l2 = F(36); P.s3l3 = F(37); P.s3l3b = F(38);
    P.s3w1 = F(39); P.s3w2 = F(40); P.s3w3 = F(41);
    P.ndbb1 = F(42); P.ndg1 = F(43); P.ndm1 = F(44); P.ndv1 = F(45);
    P.ndbb2 = F(46); P.ndg2 = F(47); P.ndm2 = F(48); P.ndv2 = F(49);
    P.ndl1 = F(50); P.ndl2 = F(51); P.ndl3 = F(52); P.ndl3b = F(53);
    P.ndw1 = F(54); P.ndw2 = F(55); P.ndw3 = F(56);
    P.xyz = F(57);
  }
}

extern "C" void kernel_launch(void* const* d_in, const int* in_sizes, int n_in,
                              void* d_out, int out_size, void* d_ws, size_t ws_size,
                              hipStream_t stream) {
  Ptrs P;
  resolve(d_in, in_sizes, n_in, P);

  // ---- workspace layout ----
  char* base = (char*)d_ws;
  size_t off = 0;
  auto alloc = [&](size_t bytes) -> char* {
    char* p = base + off;
    off += (bytes + 255) & ~(size_t)255;
    return p;
  };
  __bf16* s3w1f = (__bf16*)alloc(64 * 32 * 2);
  __bf16* s3w2f = (__bf16*)alloc(128 * 64 * 2);
  __bf16* s3w3f = (__bf16*)alloc(256 * 128 * 2);
  __bf16* ndw1f = (__bf16*)alloc(64 * 64 * 2);
  __bf16* ndw2f = (__bf16*)alloc(128 * 64 * 2);
  __bf16* ndw3f = (__bf16*)alloc(256 * 128 * 2);
  __bf16* c1af  = (__bf16*)alloc(64 * 32 * 2);
  __bf16* c1bf  = (__bf16*)alloc(64 * 64 * 2);
  __bf16* c2af  = (__bf16*)alloc(64 * 64 * 2);
  __bf16* c2bf  = (__bf16*)alloc(128 * 64 * 2);
  __bf16* c2cf  = (__bf16*)alloc(256 * 128 * 2);
  float* s3b1 = (float*)alloc(64 * 4);
  float* s3b2 = (float*)alloc(128 * 4);
  float* ndb1 = (float*)alloc(64 * 4);
  float* ndb2 = (float*)alloc(128 * 4);
  float* c1ab = (float*)alloc(64 * 4);
  float* c1bb = (float*)alloc(64 * 4);
  float* c2ab = (float*)alloc(64 * 4);
  float* c2bb = (float*)alloc(128 * 4);
  float* c2cb = (float*)alloc(256 * 4);
  unsigned* g3max  = (unsigned*)alloc(BB * 256 * 4);
  unsigned* gndmax = (unsigned*)alloc(BB * 256 * 4);
  float* T3f   = (float*)alloc(BB * 9 * 4);
  __bf16* TnBf = (__bf16*)alloc((size_t)BB * 4096 * 2);
  unsigned* bminE = (unsigned*)alloc(BB * 3 * 4);
  unsigned* bmaxE = (unsigned*)alloc(BB * 3 * 4);
  float* xyzp   = (float*)alloc((size_t)BB * 3 * NP * 4);
  __bf16* x64pm = (__bf16*)alloc((size_t)BB * NP * 64 * 2);  // point-major [B][N][64]
  unsigned* voxE = (unsigned*)alloc((size_t)BB * 256 * VV * 4);
  if (off > ws_size) return;  // workspace too small: bail deterministically

  // ---- output layout (concatenated flat, return order) ----
  float* out    = (float*)d_out;
  float* logits = out;                 // 640
  float* gfeat  = out + 640;           // 4096
  float* vact   = out + 4736;          // 16000
  int*   counts = (int*)(out + 20736); // 16000 (int32)
  int*   idxout = (int*)(out + 36736); // 262144 (int32)

  // ---- init (graph-capture-safe memsets) ----
  hipMemsetAsync(voxE, 0x00, (size_t)BB * 256 * VV * 4, stream);  // enc-space minimum
  hipMemsetAsync(g3max, 0x00, BB * 256 * 4, stream);
  hipMemsetAsync(gndmax, 0x00, BB * 256 * 4, stream);
  hipMemsetAsync(bmaxE, 0x00, BB * 3 * 4, stream);
  hipMemsetAsync(bminE, 0xFF, BB * 3 * 4, stream);
  hipMemsetAsync(counts, 0x00, BB * VV * 4, stream);

  // ---- K0: fold BN into bf16 weights ----
  auto fold = [&](const float* W, const float* g, const float* bbp, const float* m,
                  const float* v, __bf16* Wo, float* Bo, int O, int CIN, int CINP) {
    int n = O * CINP;
    k_fold<<<(n + 255) / 256, 256, 0, stream>>>(W, g, bbp, m, v, Wo, Bo, O, CIN, CINP);
  };
  fold(P.s3w1, P.s3g1, P.s3bb1, P.s3m1, P.s3v1, s3w1f, s3b1, 64, 3, 32);
  fold(P.s3w2, P.s3g2, P.s3bb2, P.s3m2, P.s3v2, s3w2f, s3b2, 128, 64, 64);
  fold(P.s3w3, nullptr, nullptr, nullptr, nullptr, s3w3f, nullptr, 256, 128, 128);
  fold(P.ndw1, P.ndg1, P.ndbb1, P.ndm1, P.ndv1, ndw1f, ndb1, 64, 64, 64);
  fold(P.ndw2, P.ndg2, P.ndbb2, P.ndm2, P.ndv2, ndw2f, ndb2, 128, 64, 64);
  fold(P.ndw3, nullptr, nullptr, nullptr, nullptr, ndw3f, nullptr, 256, 128, 128);
  fold(P.c1aw, P.c1ag, P.c1abb, P.c1am, P.c1av, c1af, c1ab, 64, 6, 32);
  fold(P.c1bw, P.c1bg, P.c1bbb, P.c1bm, P.c1bv, c1bf, c1bb, 64, 64, 64);
  fold(P.c2aw, P.c2ag, P.c2abb, P.c2am, P.c2av, c2af, c2ab, 64, 64, 64);
  fold(P.c2bw, P.c2bg, P.c2bbb, P.c2bm, P.c2bv, c2bf, c2bb, 128, 64, 64);
  fold(P.c2cw, P.c2cg, P.c2cbb, P.c2cm, P.c2cv, c2cf, c2cb, 256, 128, 128);

  dim3 gTiles(NP / TP, BB);

  // ---- pipeline ----
  k_stn3<<<gTiles, 256, 0, stream>>>(P.xyz, s3w1f, s3b1, s3w2f, s3b2, s3w3f, g3max);
  k_fc<<<BB, 256, 0, stream>>>(g3max, P.s3l1, P.s3l2, P.s3l3, P.s3l3b, T3f,
                               (__bf16*)nullptr, 9);
  k_feat<<<gTiles, 256, 0, stream>>>(P.xyz, P.features, T3f, c1af, c1ab, c1bf, c1bb,
                                     ndw1f, ndb1, ndw2f, ndb2, ndw3f,
                                     xyzp, x64pm, bminE, bmaxE, gndmax);
  k_fc<<<BB, 256, 0, stream>>>(gndmax, P.ndl1, P.ndl2, P.ndl3, P.ndl3b,
                               (float*)nullptr, TnBf, 4096);
  k_voxidx<<<dim3(NP / 256, BB), 256, 0, stream>>>(xyzp, bminE, bmaxE, idxout, counts);
  k_chain2<<<gTiles, 256, 0, stream>>>(x64pm, TnBf, c2af, c2ab, c2bf, c2bb, c2cf, c2cb,
                                       idxout, voxE);
  k_final1<<<BB, 256, 0, stream>>>(voxE, counts, P.head, logits, gfeat);
  k_vact<<<(BB * VV + 255) / 256, 256, 0, stream>>>(voxE, counts, vact);
}
